// Attention_89197880803385
// MI455X (gfx1250) — matrix operand order repaired and checked by reading, not compiled, not run on hardware
//
#include <hip/hip_runtime.h>

// ---------------------------------------------------------------------------
// MHA block for MI455X (gfx1250, wave32).  All matmuls on v_wmma_f32_16x16x32_f16.
// Prep   : x -> f16, Wqkv/Wout -> transposed f16 [N][K]   (one-time, tiny)
// Stage 1: qkvh = f16( x @ Wqkv + bqkv )                  (f16-WMMA GEMM)
// Stage 2: two-pass flash attention w/ softmax threshold  (WMMA QK^T and P.V)
// Stage 3: out  = attnh @ Wout + bout  (fp32 out)         (same GEMM kernel)
// Hot loops move only f16: LDS fills are async global->LDS b128 copies
// (gfx1250 Tensor/async path) with a synchronous v8h fallback.
// ---------------------------------------------------------------------------

typedef __attribute__((ext_vector_type(16))) _Float16 v16h;
typedef __attribute__((ext_vector_type(8)))  _Float16 v8h;
typedef __attribute__((ext_vector_type(8)))  float    v8f;
typedef __attribute__((ext_vector_type(4)))  int      v4i;

#define AS1 __attribute__((address_space(1)))
#define AS3 __attribute__((address_space(3)))

#define WMMA_F16(A, B, C) \
  __builtin_amdgcn_wmma_f32_16x16x32_f16(false, (A), false, (B), (short)0, (C), false, false)

// ---- async global->LDS copy (gfx1250) with safe synchronous fallback -------
#if defined(__gfx1250__) && \
    __has_builtin(__builtin_amdgcn_global_load_async_to_lds_b128) && \
    __has_builtin(__builtin_amdgcn_s_wait_asynccnt)
#define ASYNC_COPY 1
#else
#define ASYNC_COPY 0
#endif

__device__ __forceinline__ void copy16(_Float16* lds, const _Float16* g) {
#if ASYNC_COPY
  __builtin_amdgcn_global_load_async_to_lds_b128(
      (AS1 v4i*)g, (AS3 v4i*)lds, 0, 0);
#else
  *(v8h*)lds = *(const v8h*)g;
#endif
}
__device__ __forceinline__ void copy_fence() {
#if ASYNC_COPY
  __builtin_amdgcn_s_wait_asynccnt(0);
#endif
}

// ---- fragment loaders (CDNA5 ISA 7.12.2: 16-bit A 16x32 / B 32x16) ---------
__device__ __forceinline__ v16h cat8(v8h lo, v8h hi) {
  v16h r;
#pragma unroll
  for (int i = 0; i < 8; ++i) { r[i] = lo[i]; r[i + 8] = hi[i]; }
  return r;
}

// A fragment 16x32 (MxK): lane holds row m=lane%16, k = 16*(lane/16) + 0..15
// contiguous.  base = row 0, k-offset 0 of the strip.
__device__ __forceinline__ v16h frag_a(const _Float16* base, int stride, int lane) {
  const _Float16* p = base + (lane & 15) * stride + ((lane >> 4) << 4);
  return cat8(*(const v8h*)p, *(const v8h*)(p + 8));
}

// B fragment 32x16 (KxN) from transposed LDS tile Bt[n][k]:
// lane holds col n=lane%16, k = 16*(lane/16) + 0..15 contiguous.
__device__ __forceinline__ v16h frag_b(const _Float16* baseT, int stride, int lane) {
  const _Float16* p = baseT + (lane & 15) * stride + ((lane >> 4) << 4);
  return cat8(*(const v8h*)p, *(const v8h*)(p + 8));
}

// 16-lane butterfly row reductions (C/D fragment holds a row across lanes 0-15/16-31)
__device__ __forceinline__ void red_max16(float v[8]) {
#pragma unroll
  for (int off = 8; off; off >>= 1)
#pragma unroll
    for (int r = 0; r < 8; ++r) v[r] = fmaxf(v[r], __shfl_xor(v[r], off, 32));
}
__device__ __forceinline__ void red_add16(float v[8]) {
#pragma unroll
  for (int off = 8; off; off >>= 1)
#pragma unroll
    for (int r = 0; r < 8; ++r) v[r] += __shfl_xor(v[r], off, 32);
}

// ---------------------------------------------------------------------------
// Prep kernels (bandwidth-trivial, run once per launch)
// ---------------------------------------------------------------------------
__global__ __launch_bounds__(256) void cvt_f16(const float* __restrict__ in,
                                               _Float16* __restrict__ out, int n) {
  int i = (blockIdx.x * 256 + threadIdx.x) * 8;
  if (i >= n) return;
  float4 a = *(const float4*)(in + i);
  float4 b = *(const float4*)(in + i + 4);
  v8h o;
  o[0] = (_Float16)a.x; o[1] = (_Float16)a.y; o[2] = (_Float16)a.z; o[3] = (_Float16)a.w;
  o[4] = (_Float16)b.x; o[5] = (_Float16)b.y; o[6] = (_Float16)b.z; o[7] = (_Float16)b.w;
  *(v8h*)(out + i) = o;
}

// W[K][N] fp32 -> Wt[N][K] f16 (32x32 LDS tile, coalesced both directions)
__global__ __launch_bounds__(256) void transpose_cvt(const float* __restrict__ in,
                                                     _Float16* __restrict__ out,
                                                     int K, int N) {
  __shared__ _Float16 t[32][33];
  int k0 = blockIdx.y * 32, n0 = blockIdx.x * 32;
  int tx = threadIdx.x & 31, ty = threadIdx.x >> 5;
#pragma unroll
  for (int u = 0; u < 32; u += 8)
    t[ty + u][tx] = (_Float16)in[(size_t)(k0 + ty + u) * N + n0 + tx];
  __syncthreads();
#pragma unroll
  for (int u = 0; u < 32; u += 8)
    out[(size_t)(n0 + ty + u) * K + k0 + tx] = t[tx][ty + u];
}

// ---------------------------------------------------------------------------
// GEMM: C[M,N] = A[M,K](f16) @ BT[N,K](f16)^T + bias.   M,N %128==0, K %64==0.
// 256 threads (8 waves), tile 128x128, k-step 64 (2 WMMA k-steps per barrier).
// ---------------------------------------------------------------------------
#define GST 72  // 64 + 8 pad (rows stay 16B aligned)

template <typename OutT>
__global__ __launch_bounds__(256) void gemm_f16_wmma(
    const _Float16* __restrict__ A, const _Float16* __restrict__ BT,
    const float* __restrict__ bias, OutT* __restrict__ C,
    int M, int N, int K) {
  __shared__ _Float16 sA[128 * GST];   // [row][k]
  __shared__ _Float16 sBT[128 * GST];  // [n][k]

  const int tid = threadIdx.x, lane = tid & 31, wave = tid >> 5;
  const int m0 = (wave & 3) * 32;      // wave sub-tile: 32 rows x 64 cols
  const int n0 = (wave >> 2) * 64;
  const int tileM = blockIdx.y * 128;
  const int tileN = blockIdx.x * 128;
  const int cr = tid >> 1, cc = (tid & 1) * 32;  // copy mapping: 2 thr/row, 32 halfs

  v8f acc[2][4];
#pragma unroll
  for (int i = 0; i < 2; ++i)
#pragma unroll
    for (int j = 0; j < 4; ++j) acc[i][j] = (v8f){};

  for (int k0 = 0; k0 < K; k0 += 64) {
    const _Float16* ga = A + (size_t)(tileM + cr) * K + k0 + cc;
    const _Float16* gb = BT + (size_t)(tileN + cr) * K + k0 + cc;
    _Float16* la = sA + cr * GST + cc;
    _Float16* lb = sBT + cr * GST + cc;
#pragma unroll
    for (int u = 0; u < 4; ++u) {
      copy16(la + u * 8, ga + u * 8);
      copy16(lb + u * 8, gb + u * 8);
    }
    copy_fence();
    __syncthreads();
#pragma unroll
    for (int kt = 0; kt < 2; ++kt) {
      v16h af0 = frag_a(sA + m0 * GST + kt * 32, GST, lane);
      v16h af1 = frag_a(sA + (m0 + 16) * GST + kt * 32, GST, lane);
#pragma unroll
      for (int j = 0; j < 4; ++j) {
        v16h bf = frag_b(sBT + (n0 + j * 16) * GST + kt * 32, GST, lane);
        acc[0][j] = WMMA_F16(af0, bf, acc[0][j]);
        acc[1][j] = WMMA_F16(af1, bf, acc[1][j]);
      }
    }
    __syncthreads();
  }

  // epilogue: D layout -> lane holds col n=lane%16, rows r + 8*(lane/16)
#pragma unroll
  for (int i = 0; i < 2; ++i)
#pragma unroll
    for (int j = 0; j < 4; ++j) {
      int col = tileN + n0 + j * 16 + (lane & 15);
      float bv = bias[col];
      int rb = tileM + m0 + i * 16 + ((lane >> 4) << 3);
#pragma unroll
      for (int rr = 0; rr < 8; ++rr)
        C[(size_t)(rb + rr) * N + col] = (OutT)(acc[i][j][rr] + bv);
    }
}

// ---------------------------------------------------------------------------
// Attention per (b,h), 128 query rows per block, two-pass flash with threshold.
// qkvh rows are f16 [Q(512) | K(512) | V(512)]; per head a 64-wide slice.
// ---------------------------------------------------------------------------
#define QST 72   // 64 + 8 pad
#define VST 136  // 128 + 8 pad
#define PST 136

__global__ __launch_bounds__(256) void attn_wmma(
    const _Float16* __restrict__ qkvh, _Float16* __restrict__ outh) {
  __shared__ _Float16 sQ[128 * QST];    // [i][d]
  __shared__ _Float16 sK[128 * QST];    // [j][d]  -> B frags for Q.K^T
  __shared__ _Float16 sVT[64 * VST];    // [d][j]  -> B frags for P.V
  __shared__ _Float16 sP[8][16 * PST];  // per-wave P strip [row][j]

  const int tid = threadIdx.x, lane = tid & 31, wave = tid >> 5;
  const int b = blockIdx.y >> 3, h = blockIdx.y & 7;
  const size_t rowbase = (size_t)b * 2048;
  const int qcol = h * 64;
  const int i0 = blockIdx.x * 128;
  const float scale = 0.125f;  // 64^-0.5, applied to fp32 scores (exact)

  const int ldr = tid >> 1, ldc = (tid & 1) * 32;  // 2 thr/row, 32 halfs each

  {  // Q tile: straight f16 copy
    const _Float16* g = qkvh + (rowbase + i0 + ldr) * 1536 + qcol + ldc;
    _Float16* l = sQ + ldr * QST + ldc;
#pragma unroll
    for (int u = 0; u < 4; ++u) copy16(l + u * 8, g + u * 8);
  }
  copy_fence();
  __syncthreads();

  v16h qf0 = frag_a(sQ + wave * 16 * QST, QST, lane);
  v16h qf1 = frag_a(sQ + wave * 16 * QST + 32, QST, lane);

  float mrun[8], lrun[8];
#pragma unroll
  for (int r = 0; r < 8; ++r) { mrun[r] = -1e30f; lrun[r] = 0.f; }

  // ---- pass 1: online row max / sum over all key blocks ----
  for (int j0 = 0; j0 < 2048; j0 += 128) {
    {
      const _Float16* g = qkvh + (rowbase + j0 + ldr) * 1536 + qcol + 512 + ldc;
      _Float16* l = sK + ldr * QST + ldc;
#pragma unroll
      for (int u = 0; u < 4; ++u) copy16(l + u * 8, g + u * 8);
    }
    copy_fence();
    __syncthreads();
    v8f s[8];
#pragma unroll
    for (int jt = 0; jt < 8; ++jt) {
      v16h b0 = frag_b(sK + jt * 16 * QST, QST, lane);
      v16h b1 = frag_b(sK + jt * 16 * QST + 32, QST, lane);
      v8f c = (v8f){};
      c = WMMA_F16(qf0, b0, c);
      c = WMMA_F16(qf1, b1, c);
#pragma unroll
      for (int r = 0; r < 8; ++r) c[r] *= scale;
      s[jt] = c;
    }
    float bm[8];
#pragma unroll
    for (int r = 0; r < 8; ++r) bm[r] = s[0][r];
#pragma unroll
    for (int jt = 1; jt < 8; ++jt)
#pragma unroll
      for (int r = 0; r < 8; ++r) bm[r] = fmaxf(bm[r], s[jt][r]);
    red_max16(bm);
    float mn[8], bs[8];
#pragma unroll
    for (int r = 0; r < 8; ++r) { mn[r] = fmaxf(mrun[r], bm[r]); bs[r] = 0.f; }
#pragma unroll
    for (int jt = 0; jt < 8; ++jt)
#pragma unroll
      for (int r = 0; r < 8; ++r) bs[r] += __expf(s[jt][r] - mn[r]);
    red_add16(bs);
#pragma unroll
    for (int r = 0; r < 8; ++r) {
      lrun[r] = lrun[r] * __expf(mrun[r] - mn[r]) + bs[r];
      mrun[r] = mn[r];
    }
    __syncthreads();
  }

  float linv[8];
#pragma unroll
  for (int r = 0; r < 8; ++r) linv[r] = 1.0f / lrun[r];

  // ---- pass 2: thresholded P, accumulate O = P @ V ----
  v8f o[4];
#pragma unroll
  for (int nt = 0; nt < 4; ++nt) o[nt] = (v8f){};

  for (int j0 = 0; j0 < 2048; j0 += 128) {
    {
      const _Float16* gk = qkvh + (rowbase + j0 + ldr) * 1536 + qcol + 512 + ldc;
      const _Float16* gv = qkvh + (rowbase + j0 + ldr) * 1536 + qcol + 1024 + ldc;
      _Float16* lk = sK + ldr * QST + ldc;
#pragma unroll
      for (int u = 0; u < 4; ++u) copy16(lk + u * 8, gk + u * 8);
#pragma unroll
      for (int u = 0; u < 4; ++u) {  // V transposed into LDS (only scatter left)
        v8h w = *(const v8h*)(gv + u * 8);
#pragma unroll
        for (int e = 0; e < 8; ++e) sVT[(ldc + u * 8 + e) * VST + ldr] = w[e];
      }
    }
    copy_fence();
    __syncthreads();
#pragma unroll
    for (int jt = 0; jt < 8; ++jt) {
      v16h b0 = frag_b(sK + jt * 16 * QST, QST, lane);
      v16h b1 = frag_b(sK + jt * 16 * QST + 32, QST, lane);
      v8f c = (v8f){};
      c = WMMA_F16(qf0, b0, c);
      c = WMMA_F16(qf1, b1, c);
      // p = softmax, thresholded; write D-layout -> LDS row-major strip
#pragma unroll
      for (int r = 0; r < 8; ++r) {
        float p = __expf(c[r] * scale - mrun[r]) * linv[r];
        p = (p > 0.01f) ? p : 0.0f;
        sP[wave][(r + ((lane >> 4) << 3)) * PST + jt * 16 + (lane & 15)] = (_Float16)p;
      }
    }
    __syncthreads();
#pragma unroll
    for (int kt = 0; kt < 4; ++kt) {
      v16h pa = frag_a(sP[wave] + kt * 32, PST, lane);
#pragma unroll
      for (int nt = 0; nt < 4; ++nt) {
        v16h vb = frag_b(sVT + nt * 16 * VST + kt * 32, VST, lane);
        o[nt] = WMMA_F16(pa, vb, o[nt]);
      }
    }
    __syncthreads();
  }

  // store attention output f16 as [b*2048 + i][h*64 + d] (input of final GEMM)
#pragma unroll
  for (int nt = 0; nt < 4; ++nt) {
    int col = qcol + nt * 16 + (lane & 15);
    int rb = i0 + wave * 16 + ((lane >> 4) << 3);
#pragma unroll
    for (int rr = 0; rr < 8; ++rr)
      outh[(rowbase + rb + rr) * 512 + col] = (_Float16)o[nt][rr];
  }
}

// ---------------------------------------------------------------------------

extern "C" void kernel_launch(void* const* d_in, const int* in_sizes, int n_in,
                              void* d_out, int out_size, void* d_ws, size_t ws_size,
                              hipStream_t stream) {
  (void)in_sizes; (void)n_in; (void)out_size; (void)ws_size;
  const float* x    = (const float*)d_in[0];   // [4,2048,512]
  const float* Wqkv = (const float*)d_in[1];   // [512,1536]
  const float* bqkv = (const float*)d_in[2];   // [1536]
  const float* Wout = (const float*)d_in[3];   // [512,512]
  const float* bout = (const float*)d_in[4];   // [512]
  float* out = (float*)d_out;                  // [4,2048,512]

  char* ws = (char*)d_ws;
  _Float16* xh     = (_Float16*)(ws);                           //  8 MB
  _Float16* WqkvT  = (_Float16*)(ws + (size_t)8  * 1024 * 1024);// 1.5 MB [1536][512]
  _Float16* WoutT  = (_Float16*)(ws + (size_t)10 * 1024 * 1024);// 0.5 MB [512][512]
  _Float16* qkvh   = (_Float16*)(ws + (size_t)11 * 1024 * 1024);// 24 MB  [8192][1536]
  _Float16* attnh  = (_Float16*)(ws + (size_t)36 * 1024 * 1024);//  8 MB  [8192][512]

  dim3 blk(256);

  cvt_f16<<<8192 * 512 / (256 * 8), blk, 0, stream>>>(x, xh, 8192 * 512);
  transpose_cvt<<<dim3(1536 / 32, 512 / 32), blk, 0, stream>>>(Wqkv, WqkvT, 512, 1536);
  transpose_cvt<<<dim3(512 / 32, 512 / 32), blk, 0, stream>>>(Wout, WoutT, 512, 512);

  gemm_f16_wmma<_Float16><<<dim3(1536 / 128, 8192 / 128), blk, 0, stream>>>(
      xh, WqkvT, bqkv, qkvh, 8192, 1536, 512);

  attn_wmma<<<dim3(2048 / 128, 32), blk, 0, stream>>>(qkvh, attnh);

  gemm_f16_wmma<float><<<dim3(512 / 128, 8192 / 128), blk, 0, stream>>>(
      attnh, WoutT, bout, out, 8192, 512, 512);
}
